// DeepFM_14594298872614
// MI455X (gfx1250) — matrix operand order repaired and checked by reading, not compiled, not run on hardware
//
#include <hip/hip_runtime.h>
#include <hip/hip_bf16.h>

// ---------------------------------------------------------------------------
// DeepFM forward on gfx1250: bf16 WMMA (v_wmma_f32_16x16x32_bf16) everywhere.
// Round 3: 32x64 output tile per wave in the GEMM (B-operand reuse across two
// M-tiles -> 1.5 vmem loads per WMMA instead of 2.5).
// ---------------------------------------------------------------------------

typedef __attribute__((ext_vector_type(16))) __bf16 v16bf;
typedef __attribute__((ext_vector_type(8)))  float  v8f;
typedef __attribute__((ext_vector_type(4)))  __bf16 v4bf;

union BF16x16 {           // one WMMA A/B operand (16 halves per lane = 32B)
  v16bf v;
  uint4 q[2];
};

static constexpr int BSZ   = 16384;
static constexpr int F     = 64;
static constexpr int D     = 32;
static constexpr int H     = 256;
static constexpr int HO    = 128;
static constexpr int NPAIR = F * (F - 1) / 2;   // 2016
static constexpr int CIN   = HO + NPAIR + D;    // 2176
static constexpr int KX    = F * D;             // 2048

// ---------------- workspace layout (bytes, all offsets 256-aligned) --------
static constexpr size_t OFF_XB   = 0;                                   // B*KX bf16
static constexpr size_t OFF_W1T  = OFF_XB   + (size_t)BSZ * KX * 2;     // [H][KX]
static constexpr size_t OFF_W2T  = OFF_W1T  + (size_t)H * KX * 2;       // [H][H]
static constexpr size_t OFF_W3T  = OFF_W2T  + (size_t)H * H * 2;        // [HO][H]
static constexpr size_t OFF_WC1T = OFF_W3T  + (size_t)HO * H * 2;       // [H][CIN]
static constexpr size_t OFF_H1   = OFF_WC1T + (size_t)H * CIN * 2;      // [B][H]
static constexpr size_t OFF_H2   = OFF_H1   + (size_t)BSZ * H * 2;      // [B][H]
static constexpr size_t OFF_CIN  = OFF_H2   + (size_t)BSZ * H * 2;      // [B][CIN]
static constexpr size_t OFF_C    = OFF_CIN  + (size_t)BSZ * CIN * 2;    // [B][H]

// ---------------- elementwise converts -------------------------------------
__global__ __launch_bounds__(256) void cvt_f32_bf16_vec4(
    const float* __restrict__ src, __bf16* __restrict__ dst, long n4) {
  long t = (long)blockIdx.x * blockDim.x + threadIdx.x;
  if (t >= n4) return;
  float4 v = ((const float4*)src)[t];
  v4bf o;
  o[0] = (__bf16)v.x; o[1] = (__bf16)v.y; o[2] = (__bf16)v.z; o[3] = (__bf16)v.w;
  ((v4bf*)dst)[t] = o;
}

// dst[n*K + k] = (bf16) src[k*N + n]   (weights: [K][N] f32 -> [N][K] bf16)
__global__ __launch_bounds__(256) void cvt_transpose_bf16(
    const float* __restrict__ src, __bf16* __restrict__ dst, int K, int N) {
  int t = blockIdx.x * blockDim.x + threadIdx.x;
  if (t >= K * N) return;
  int n = t / K, k = t - n * K;
  dst[t] = (__bf16)src[(long)k * N + n];
}

// ---------------- activation ------------------------------------------------
__device__ __forceinline__ float act_apply(float z, int act) {
  if (act == 1) {                                   // SELU
    const float lam = 1.0507009873554805f;
    const float la  = 1.7580993408473766f;          // lam * alpha
    return z > 0.f ? lam * z : la * (__expf(z) - 1.f);
  }
  if (act == 2) return z > 0.f ? z : 0.f;           // ReLU
  return z;
}

// upper-triangular pair index for (i,j), i<j, F=64
__device__ __forceinline__ int triu_idx(int i, int j) {
  return (i * (2 * F - i - 1)) / 2 + (j - i - 1);
}

// ---------------- gram / soi / first_order ---------------------------------
// 1 wave per batch row: gram(b) = x_b (64x32) * x_b^T via 10 WMMAs (K=32=D).
// Off-diagonal tiles (ti<tj): every element is in the triu -> unconditional
// stores (no exec-mask branching). Diagonal tiles: predicated on j>i.
__global__ __launch_bounds__(128) void gram_soi_first(
    const __bf16* __restrict__ xb, const float* __restrict__ xf,
    __bf16* __restrict__ cin) {
  const int lane = threadIdx.x & 31;
  const int wave = threadIdx.x >> 5;
  const int b    = blockIdx.x * 4 + wave;
  const int mr   = lane & 15;
  const int hi   = lane >> 4;

  const __bf16* xrow = xb + (long)b * KX;
  BF16x16 a[4], bm[4];
#pragma unroll
  for (int g = 0; g < 4; ++g) {
    const __bf16* r = xrow + (g * 16 + mr) * D;
    // A operand: row mr, halves [c0,c0+8) and [c0+16,c0+24), c0 = 8*hi
    a[g].q[0]  = *(const uint4*)(r + hi * 8);
    a[g].q[1]  = *(const uint4*)(r + hi * 8 + 16);
    // B operand (x^T): column mr = row mr of x, halves [c0,c0+8) and [c0+16,c0+24)
    bm[g].q[0] = *(const uint4*)(r + hi * 8);
    bm[g].q[1] = *(const uint4*)(r + hi * 8 + 16);
  }

  __bf16* soi = cin + (long)b * CIN + HO;

  // ---- off-diagonal tiles: branch-free scatter -----------------------------
#pragma unroll
  for (int ti = 0; ti < 4; ++ti) {
#pragma unroll
    for (int tj = ti + 1; tj < 4; ++tj) {
      v8f c = {};
      c = __builtin_amdgcn_wmma_f32_16x16x32_bf16(
          false, a[ti].v, false, bm[tj].v, (short)0, c, false, false);
      const int j = tj * 16 + mr;
#pragma unroll
      for (int r8 = 0; r8 < 8; ++r8) {
        const int i = ti * 16 + r8 + 8 * hi;   // i < ti*16+16 <= tj*16 <= j
        soi[triu_idx(i, j)] = (__bf16)c[r8];
      }
    }
  }

  // ---- diagonal tiles: predicated scatter ----------------------------------
#pragma unroll
  for (int ti = 0; ti < 4; ++ti) {
    v8f c = {};
    c = __builtin_amdgcn_wmma_f32_16x16x32_bf16(
        false, a[ti].v, false, bm[ti].v, (short)0, c, false, false);
    const int j = ti * 16 + mr;
#pragma unroll
    for (int r8 = 0; r8 < 8; ++r8) {
      const int i = ti * 16 + r8 + 8 * hi;
      if (j > i) soi[triu_idx(i, j)] = (__bf16)c[r8];
    }
  }

  // first_order: sum over F rows of the f32 input (coalesced per lane)
  float s = 0.f;
  const float* xfb = xf + (long)b * KX;
  for (int i = 0; i < F; ++i) s += xfb[i * D + lane];
  cin[(long)b * CIN + HO + NPAIR + lane] = (__bf16)s;
}

// ---------------- generic WMMA GEMM: D = act(A[M,K] * BT[N,K]^T + bias) ----
// Each wave computes a 32x64 tile (2 M-tiles x 4 N-tiles): each B operand is
// reused by two WMMAs. grid.x = M/128 (4 waves/block, 32 rows each),
// grid.y = N/64.
__global__ __launch_bounds__(128) void gemm_wmma_bf16(
    const __bf16* __restrict__ A, long ldA,
    const __bf16* __restrict__ BT, int K,
    const float* __restrict__ bias,
    __bf16* __restrict__ Dst, long ldD, int act) {
  const int lane = threadIdx.x & 31;
  const int wave = threadIdx.x >> 5;
  const int m0   = (blockIdx.x * 4 + wave) * 32;
  const int n0   = blockIdx.y * 64;
  const int mr   = lane & 15;
  const int hi   = lane >> 4;

  v8f acc0[4] = {};            // rows m0 .. m0+15
  v8f acc1[4] = {};            // rows m0+16 .. m0+31
  const __bf16* arow0 = A + (long)(m0 + mr) * ldA;
  const __bf16* arow1 = arow0 + (long)16 * ldA;
  const int c0 = hi * 8;    // A chunk base
  const int k0 = hi * 8;    // B chunk base

  for (int k = 0; k < K; k += 32) {
    BF16x16 a0, a1;
    a0.q[0] = *(const uint4*)(arow0 + k + c0);
    a0.q[1] = *(const uint4*)(arow0 + k + c0 + 16);
    a1.q[0] = *(const uint4*)(arow1 + k + c0);
    a1.q[1] = *(const uint4*)(arow1 + k + c0 + 16);
#pragma unroll
    for (int t = 0; t < 4; ++t) {
      const __bf16* brow = BT + (long)(n0 + t * 16 + mr) * K + k + k0;
      BF16x16 bm;
      bm.q[0] = *(const uint4*)(brow);
      bm.q[1] = *(const uint4*)(brow + 16);
      acc0[t] = __builtin_amdgcn_wmma_f32_16x16x32_bf16(
          false, a0.v, false, bm.v, (short)0, acc0[t], false, false);
      acc1[t] = __builtin_amdgcn_wmma_f32_16x16x32_bf16(
          false, a1.v, false, bm.v, (short)0, acc1[t], false, false);
    }
  }

#pragma unroll
  for (int t = 0; t < 4; ++t) {
    const int n  = n0 + t * 16 + mr;
    const float bv = bias ? bias[n] : 0.f;
#pragma unroll
    for (int r = 0; r < 8; ++r) {
      const int m = m0 + r + 8 * hi;
      Dst[(long)m * ldD + n]        = (__bf16)act_apply(acc0[t][r] + bv, act);
      Dst[(long)(m + 16) * ldD + n] = (__bf16)act_apply(acc1[t][r] + bv, act);
    }
  }
}

// ---------------- final: out[b] = c[b,:] . Wc2 + bc2 -----------------------
__global__ __launch_bounds__(256) void final_dot(
    const __bf16* __restrict__ c, const float* __restrict__ Wc2,
    const float* __restrict__ bc2, float* __restrict__ out) {
  const int lane = threadIdx.x & 31;
  const int wave = threadIdx.x >> 5;
  const int row  = blockIdx.x * 8 + wave;
  float s = 0.f;
  const __bf16* cr = c + (long)row * H;
#pragma unroll
  for (int k = lane; k < H; k += 32) s += (float)cr[k] * Wc2[k];
#pragma unroll
  for (int off = 16; off; off >>= 1) s += __shfl_xor(s, off, 32);
  if (lane == 0) out[row] = s + bc2[0];
}

// ---------------------------------------------------------------------------
extern "C" void kernel_launch(void* const* d_in, const int* in_sizes, int n_in,
                              void* d_out, int out_size, void* d_ws, size_t ws_size,
                              hipStream_t stream) {
  const float* x   = (const float*)d_in[0];
  const float* W1  = (const float*)d_in[1];
  const float* b1  = (const float*)d_in[2];
  const float* W2  = (const float*)d_in[3];
  const float* b2  = (const float*)d_in[4];
  const float* W3  = (const float*)d_in[5];
  const float* b3  = (const float*)d_in[6];
  const float* Wc1 = (const float*)d_in[7];
  const float* bc1 = (const float*)d_in[8];
  const float* Wc2 = (const float*)d_in[9];
  const float* bc2 = (const float*)d_in[10];
  float* out = (float*)d_out;

  char* ws = (char*)d_ws;
  __bf16* xb   = (__bf16*)(ws + OFF_XB);
  __bf16* W1T  = (__bf16*)(ws + OFF_W1T);
  __bf16* W2T  = (__bf16*)(ws + OFF_W2T);
  __bf16* W3T  = (__bf16*)(ws + OFF_W3T);
  __bf16* Wc1T = (__bf16*)(ws + OFF_WC1T);
  __bf16* h1   = (__bf16*)(ws + OFF_H1);
  __bf16* h2   = (__bf16*)(ws + OFF_H2);
  __bf16* cin  = (__bf16*)(ws + OFF_CIN);
  __bf16* cbuf = (__bf16*)(ws + OFF_C);

  // 1) precision conversions
  {
    long n4 = (long)BSZ * KX / 4;
    cvt_f32_bf16_vec4<<<dim3((unsigned)((n4 + 255) / 256)), dim3(256), 0, stream>>>(x, xb, n4);
  }
  cvt_transpose_bf16<<<dim3((KX * H + 255) / 256), dim3(256), 0, stream>>>(W1, W1T, KX, H);
  cvt_transpose_bf16<<<dim3((H * H + 255) / 256),  dim3(256), 0, stream>>>(W2, W2T, H, H);
  cvt_transpose_bf16<<<dim3((H * HO + 255) / 256), dim3(256), 0, stream>>>(W3, W3T, H, HO);
  cvt_transpose_bf16<<<dim3((CIN * H + 255) / 256), dim3(256), 0, stream>>>(Wc1, Wc1T, CIN, H);

  // 2) gram -> soi + first_order  (10 WMMAs per batch row)
  gram_soi_first<<<dim3(BSZ / 4), dim3(128), 0, stream>>>(xb, x, cin);

  // 3) MLP: h1 = selu(x@W1+b1); h2 = selu(h1@W2+b2); hoi = h2@W3+b3 -> cin[:,0:128]
  gemm_wmma_bf16<<<dim3(BSZ / 128, H / 64), dim3(128), 0, stream>>>(
      xb, (long)KX, W1T, KX, b1, h1, (long)H, 1);
  gemm_wmma_bf16<<<dim3(BSZ / 128, H / 64), dim3(128), 0, stream>>>(
      h1, (long)H, W2T, H, b2, h2, (long)H, 1);
  gemm_wmma_bf16<<<dim3(BSZ / 128, HO / 64), dim3(128), 0, stream>>>(
      h2, (long)H, W3T, H, b3, cin, (long)CIN, 0);

  // 4) classifier: c = relu(cin@Wc1+bc1)
  gemm_wmma_bf16<<<dim3(BSZ / 128, H / 64), dim3(128), 0, stream>>>(
      cin, (long)CIN, Wc1T, CIN, bc1, cbuf, (long)H, 2);

  // 5) out = c@Wc2 + bc2
  final_dot<<<dim3(BSZ / 8), dim3(256), 0, stream>>>(cbuf, Wc2, bc2, out);

  (void)in_sizes; (void)n_in; (void)out_size; (void)ws_size;
}
